// BidirectionalCrossAttention_10711648436564
// MI455X (gfx1250) — compile-verified, matrix operand order checked by reading, not run on hardware
//
#include <hip/hip_runtime.h>
#include <hip/hip_bf16.h>

typedef __attribute__((ext_vector_type(16))) __bf16 v16bf;
typedef __attribute__((ext_vector_type(8)))  float  v8f;
typedef __attribute__((ext_vector_type(4)))  unsigned int u32x4;

// ---------- helpers ----------

__device__ __forceinline__ unsigned short f2bf(float f) {
    union { float f; unsigned u; } c; c.f = f;
    unsigned u = c.u;
    u += 0x7fffu + ((u >> 16) & 1u);      // round-to-nearest-even
    return (unsigned short)(u >> 16);
}

__device__ __forceinline__ v16bf make_frag(u32x4 lo, u32x4 hi) {
    union { u32x4 q[2]; v16bf f; } cv;
    cv.q[0] = lo; cv.q[1] = hi;
    return cv.f;
}

// 16x32 bf16 A/B fragment from row-major matrix: lane L<16 holds row L,
// K {0..7,16..23}; lane L+16 holds row L, K {8..15,24..31}. Two 16B reads.
__device__ __forceinline__ v16bf lds_frag(const unsigned short* base, int row0, int lda, int kcol) {
    const int lane = threadIdx.x & 31;
    const int r  = lane & 15;
    const int kb = (lane >> 4) << 3;
    const unsigned short* p = base + (size_t)(row0 + r) * lda + kcol + kb;
    u32x4 lo = *(const u32x4*)p;
    u32x4 hi = *(const u32x4*)(p + 16);
    return make_frag(lo, hi);
}

__device__ __forceinline__ v16bf gmem_frag(const unsigned short* rowp) {
    u32x4 lo = *(const u32x4*)rowp;
    u32x4 hi = *(const u32x4*)(rowp + 16);
    return make_frag(lo, hi);
}

__device__ __forceinline__ v8f wmma_bf16(v16bf a, v16bf b, v8f c) {
    return __builtin_amdgcn_wmma_f32_16x16x32_bf16(false, a, false, b, (short)0, c, false, false);
}

__device__ __forceinline__ v8f zero8() {
    v8f z;
#pragma unroll
    for (int e = 0; e < 8; ++e) z[e] = 0.0f;
    return z;
}

// max-reduce across the 16-lane half-groups (rows live per 16 lanes).
// Prefer v_permlane16_b32 (VALU) over ds_bpermute (DS unit).
__device__ __forceinline__ float maxred16(float v) {
#if __has_builtin(__builtin_amdgcn_permlane16)
    unsigned x, o;
    x = __float_as_uint(v);
    o = __builtin_amdgcn_permlane16(x, x, 0x67452301u, 0xEFCDAB89u, false, false); // xor 1
    v = fmaxf(v, __uint_as_float(o)); x = __float_as_uint(v);
    o = __builtin_amdgcn_permlane16(x, x, 0x54761032u, 0xDCFE98BAu, false, false); // xor 2
    v = fmaxf(v, __uint_as_float(o)); x = __float_as_uint(v);
    o = __builtin_amdgcn_permlane16(x, x, 0x32107654u, 0xBA98FEDCu, false, false); // xor 4
    v = fmaxf(v, __uint_as_float(o)); x = __float_as_uint(v);
    o = __builtin_amdgcn_permlane16(x, x, 0xFEDCBA98u, 0x76543210u, false, false); // xor 8
    v = fmaxf(v, __uint_as_float(o));
    return v;
#else
    v = fmaxf(v, __shfl_xor(v, 1));
    v = fmaxf(v, __shfl_xor(v, 2));
    v = fmaxf(v, __shfl_xor(v, 4));
    v = fmaxf(v, __shfl_xor(v, 8));
    return v;
#endif
}

// ---------- CDNA5 async global->LDS (ASYNCcnt path) via inline asm ----------
// The clang builtin's pointer LangAS typing is toolchain-specific; inline asm
// bypasses it (recommended portable path). LDS address operand = low 32 bits
// of a generic pointer to a __shared__ object (the LDS byte offset).
__device__ __forceinline__ void async16(const unsigned short* g, unsigned short* l) {
    unsigned lds = (unsigned)(size_t)(void*)l;
    asm volatile("global_load_async_to_lds_b128 %0, %1, off"
                 :: "v"(lds), "v"(g)
                 : "memory");
}
__device__ __forceinline__ void wait_async0() {
    asm volatile("s_wait_asynccnt 0" ::: "memory");
}

// ---------- GEMM: C = A @ W^T (+bias), A [rows,K] (f32 or bf16), W [N,K] f32 ----------
// Block 256 thr = 8 waves; tile 128x128x32; wave computes 32x64.
// Double-buffered LDS with register prefetch: one barrier per k-step.

template<bool A_BF16, bool WRITE_BF16, bool HAS_BIAS>
__global__ __launch_bounds__(256)
void gemm_tn(const void* __restrict__ Ap, const float* __restrict__ W,
             const float* __restrict__ bias, void* __restrict__ Cp,
             int K, int ldc, int col_off)
{
    __shared__ unsigned short sA[2][128 * 40];
    __shared__ unsigned short sW[2][128 * 40];

    const int tid  = threadIdx.x;
    const int lane = tid & 31;
    const int wave = tid >> 5;
    const int wr = wave >> 1;          // 0..3
    const int wc = wave & 1;           // 0..1
    const int rowBase = blockIdx.y * 128;
    const int colBase = blockIdx.x * 128;

    float4 aF[4]; u32x4 aH[2]; float4 wF[4];

    auto loadTile = [&](int k0) {
        if (A_BF16) {
            const unsigned short* A = (const unsigned short*)Ap;
#pragma unroll
            for (int i = 0; i < 2; ++i) {
                int v = tid + i * 256; int row = v >> 2, c8 = v & 3;
                aH[i] = *(const u32x4*)(A + (size_t)(rowBase + row) * K + k0 + c8 * 8);
            }
        } else {
            const float* A = (const float*)Ap;
#pragma unroll
            for (int i = 0; i < 4; ++i) {
                int v = tid + i * 256; int row = v >> 3, c4 = v & 7;
                aF[i] = *(const float4*)(A + (size_t)(rowBase + row) * K + k0 + c4 * 4);
            }
        }
#pragma unroll
        for (int i = 0; i < 4; ++i) {
            int v = tid + i * 256; int row = v >> 3, c4 = v & 7;
            wF[i] = *(const float4*)(W + (size_t)(colBase + row) * K + k0 + c4 * 4);
        }
    };
    auto stageTile = [&](int buf) {
        if (A_BF16) {
#pragma unroll
            for (int i = 0; i < 2; ++i) {
                int v = tid + i * 256; int row = v >> 2, c8 = v & 3;
                *(u32x4*)&sA[buf][row * 40 + c8 * 8] = aH[i];
            }
        } else {
#pragma unroll
            for (int i = 0; i < 4; ++i) {
                int v = tid + i * 256; int row = v >> 3, c4 = v & 7;
                ushort4 hv; hv.x = f2bf(aF[i].x); hv.y = f2bf(aF[i].y);
                hv.z = f2bf(aF[i].z); hv.w = f2bf(aF[i].w);
                *(ushort4*)&sA[buf][row * 40 + c4 * 4] = hv;
            }
        }
#pragma unroll
        for (int i = 0; i < 4; ++i) {
            int v = tid + i * 256; int row = v >> 3, c4 = v & 7;
            ushort4 hv; hv.x = f2bf(wF[i].x); hv.y = f2bf(wF[i].y);
            hv.z = f2bf(wF[i].z); hv.w = f2bf(wF[i].w);
            *(ushort4*)&sW[buf][row * 40 + c4 * 4] = hv;
        }
    };

    v8f acc[2][4];
#pragma unroll
    for (int i = 0; i < 2; ++i)
#pragma unroll
        for (int j = 0; j < 4; ++j) acc[i][j] = zero8();

    loadTile(0); stageTile(0);
    __syncthreads();
    int cur = 0;
    for (int k0 = 0; k0 < K; k0 += 32) {
        const bool hn = (k0 + 32) < K;
        if (hn) loadTile(k0 + 32);                     // global -> regs, overlaps wmma

        v16bf af[2], bfm[4];
#pragma unroll
        for (int i = 0; i < 2; ++i) af[i] = lds_frag(&sA[cur][0], wr * 32 + i * 16, 40, 0);
#pragma unroll
        for (int j = 0; j < 4; ++j) bfm[j] = lds_frag(&sW[cur][0], wc * 64 + j * 16, 40, 0);
#pragma unroll
        for (int i = 0; i < 2; ++i)
#pragma unroll
            for (int j = 0; j < 4; ++j) acc[i][j] = wmma_bf16(af[i], bfm[j], acc[i][j]);

        if (hn) stageTile(cur ^ 1);                    // regs -> other LDS buffer
        __syncthreads();
        cur ^= 1;
    }

    const int rb = (lane >> 4) << 3;
    const int cc = lane & 15;
#pragma unroll
    for (int i = 0; i < 2; ++i)
#pragma unroll
        for (int j = 0; j < 4; ++j) {
            int gr = rowBase + wr * 32 + i * 16 + rb;
            int gc = colBase + wc * 64 + j * 16 + cc;
            float bv = HAS_BIAS ? bias[gc] : 0.0f;
#pragma unroll
            for (int e = 0; e < 8; ++e) {
                float vv = acc[i][j][e] + bv;
                if (WRITE_BF16)
                    ((unsigned short*)Cp)[(size_t)(gr + e) * ldc + col_off + gc] = f2bf(vv);
                else
                    ((float*)Cp)[(size_t)(gr + e) * ldc + col_off + gc] = vv;
            }
        }
}

// ---------- Flash attention pass (both directions by swapping args) ----------
// Q,K: bf16 [B*n,1536], head h at col h*128. V/O: bf16 [B*n,768], head h at col h*64.
// Grid (nQ/64, H, B), block 128 = 4 waves, wave owns 16 Q rows.
// K tile: async global->LDS (double buffered). V tile: reg-prefetched + transposed.

__global__ __launch_bounds__(128)
void flash_bidir(const unsigned short* __restrict__ Q,
                 const unsigned short* __restrict__ Km,
                 const unsigned short* __restrict__ V,
                 unsigned short* __restrict__ O,
                 int nQ, int nK, float scale)
{
    const int b = blockIdx.z, h = blockIdx.y;
    const int tid = threadIdx.x;
    const int wave = tid >> 5, lane = tid & 31;
    const int q0 = blockIdx.x * 64;
    const int r  = lane & 15;
    const int kb = (lane >> 4) << 3;
    const int rb = kb;

    __shared__ unsigned short sK[2][32 * 136];   // 32 keys x 128 dims (pad 8)
    __shared__ unsigned short sVT[2][64 * 40];   // V^T: 64 outdims x 32 keys (pad 8)
    __shared__ unsigned short sP[4][16 * 40];    // per-wave P tile 16x32 (pad 8)

    const unsigned short* qrowp = Q + ((size_t)(b * nQ + q0 + wave * 16 + r)) * 1536 + h * 128 + kb;
    v16bf qf[4];
#pragma unroll
    for (int d = 0; d < 4; ++d) qf[d] = gmem_frag(qrowp + d * 32);

    u32x4 ou; ou[0] = ou[1] = ou[2] = ou[3] = 0x3F803F80u;   // bf16 1.0 x16
    const v16bf onesf = make_frag(ou, ou);

    v8f acc[4];
#pragma unroll
    for (int n = 0; n < 4; ++n) acc[n] = zero8();
    float mrow[8], lrow[8];
#pragma unroll
    for (int j = 0; j < 8; ++j) { mrow[j] = -1e30f; lrow[j] = 0.0f; }

    auto issueK = [&](int kt, int buf) {           // 32x128 bf16, 16B per thread x4
#pragma unroll
        for (int i = 0; i < 4; ++i) {
            int v = tid + i * 128; int row = v >> 4, c8 = v & 15;
            const unsigned short* g = Km + ((size_t)(b * nK + kt + row)) * 1536 + h * 128 + c8 * 8;
            async16(g, &sK[buf][row * 136 + c8 * 8]);
        }
    };
    auto loadV = [&](int kt, ushort4* vr) {
#pragma unroll
        for (int i = 0; i < 4; ++i) {
            int v = tid + i * 128; int row = v >> 4, c4 = v & 15;
            vr[i] = *(const ushort4*)(V + ((size_t)(b * nK + kt + row)) * 768 + h * 64 + c4 * 4);
        }
    };
    auto stageVT = [&](const ushort4* vr, int buf) {
#pragma unroll
        for (int i = 0; i < 4; ++i) {
            int v = tid + i * 128; int row = v >> 4, c4 = v & 15;
            sVT[buf][(c4 * 4 + 0) * 40 + row] = vr[i].x;
            sVT[buf][(c4 * 4 + 1) * 40 + row] = vr[i].y;
            sVT[buf][(c4 * 4 + 2) * 40 + row] = vr[i].z;
            sVT[buf][(c4 * 4 + 3) * 40 + row] = vr[i].w;
        }
    };

    // prologue: tile 0
    {
        ushort4 vr[4];
        issueK(0, 0);
        loadV(0, vr);
        stageVT(vr, 0);
        wait_async0();
    }
    __syncthreads();

    int cur = 0;
    for (int kt = 0; kt < nK; kt += 32) {
        const int nxt = cur ^ 1;
        const bool hn = (kt + 32) < nK;
        ushort4 vr[4];
        if (hn) {
            issueK(kt + 32, nxt);                  // async bypasses VGPRs
            loadV(kt + 32, vr);                    // V prefetch to regs
        }

        // S = (Q @ K^T) * scale -> 16x32 f32
        v8f S0 = zero8(), S1 = zero8();
#pragma unroll
        for (int d = 0; d < 4; ++d) S0 = wmma_bf16(qf[d], lds_frag(&sK[cur][0], 0, 136, d * 32), S0);
#pragma unroll
        for (int d = 0; d < 4; ++d) S1 = wmma_bf16(qf[d], lds_frag(&sK[cur][0], 16, 136, d * 32), S1);

        float mnew[8];
#pragma unroll
        for (int j = 0; j < 8; ++j) {
            S0[j] *= scale; S1[j] *= scale;
            mnew[j] = fmaxf(mrow[j], maxred16(fmaxf(S0[j], S1[j])));
        }
#pragma unroll
        for (int j = 0; j < 8; ++j) {
            float corr = __expf(mrow[j] - mnew[j]);
            lrow[j] *= corr;
            mrow[j]  = mnew[j];
#pragma unroll
            for (int n = 0; n < 4; ++n) acc[n][j] *= corr;
        }
        unsigned short* Pw = &sP[wave][0];
#pragma unroll
        for (int j = 0; j < 8; ++j) {
            S0[j] = __expf(S0[j] - mnew[j]);
            S1[j] = __expf(S1[j] - mnew[j]);
            Pw[(rb + j) * 40 + r]      = f2bf(S0[j]);
            Pw[(rb + j) * 40 + 16 + r] = f2bf(S1[j]);
        }
        asm volatile("s_wait_dscnt 0" ::: "memory");   // P write -> P read (same wave)

        v16bf pf = lds_frag(Pw, 0, 40, 0);
        v8f rs = wmma_bf16(pf, onesf, zero8());        // row sums of P via WMMA
#pragma unroll
        for (int n = 0; n < 4; ++n)
            acc[n] = wmma_bf16(pf, lds_frag(&sVT[cur][0], n * 16, 40, 0), acc[n]);
#pragma unroll
        for (int j = 0; j < 8; ++j) lrow[j] += rs[j];

        if (hn) {
            stageVT(vr, nxt);
            wait_async0();                         // next K tile resident
        }
        __syncthreads();                           // one barrier per key tile
        cur = nxt;
    }

    float rinv[8];
#pragma unroll
    for (int j = 0; j < 8; ++j) rinv[j] = 1.0f / lrow[j];
#pragma unroll
    for (int n = 0; n < 4; ++n)
#pragma unroll
        for (int j = 0; j < 8; ++j) {
            int orow = b * nQ + q0 + wave * 16 + rb + j;
            O[(size_t)orow * 768 + h * 64 + n * 16 + r] = f2bf(acc[n][j] * rinv[j]);
        }
}

// ---------- launch ----------

extern "C" void kernel_launch(void* const* d_in, const int* in_sizes, int n_in,
                              void* d_out, int out_size, void* d_ws, size_t ws_size,
                              hipStream_t stream) {
    const float* x          = (const float*)d_in[0];
    const float* src        = (const float*)d_in[1];
    const float* qk_w       = (const float*)d_in[2];
    const float* qk_src_w   = (const float*)d_in[3];
    const float* v_w        = (const float*)d_in[4];
    const float* v_src_w    = (const float*)d_in[5];
    const float* proj_w     = (const float*)d_in[6];
    const float* proj_b     = (const float*)d_in[7];
    const float* proj_src_w = (const float*)d_in[8];
    const float* proj_src_b = (const float*)d_in[9];
    float* out = (float*)d_out;

    const int B = 2, N = 2048, M = 2048, C = 768, D2 = 1536, Dd = 768, H = 12;
    const float scale = 0.125f;   // 64^-0.5

    char* ws = (char*)d_ws;
    size_t off = 0;
    auto alloc = [&](size_t bytes) { char* p = ws + off; off += (bytes + 255) & ~(size_t)255; return p; };
    unsigned short* qk  = (unsigned short*)alloc((size_t)B * N * D2 * 2);
    unsigned short* qks = (unsigned short*)alloc((size_t)B * M * D2 * 2);
    unsigned short* v   = (unsigned short*)alloc((size_t)B * N * Dd * 2);
    unsigned short* vs  = (unsigned short*)alloc((size_t)B * M * Dd * 2);
    unsigned short* y   = (unsigned short*)alloc((size_t)B * N * Dd * 2);
    unsigned short* ys  = (unsigned short*)alloc((size_t)B * M * Dd * 2);
    (void)in_sizes; (void)n_in; (void)out_size; (void)ws_size;

    dim3 blk(256);
    // Stage 1: input projections -> bf16
    gemm_tn<false, true, false><<<dim3(D2 / 128, B * N / 128), blk, 0, stream>>>(x,   qk_w,     nullptr, qk,  C, D2, 0);
    gemm_tn<false, true, false><<<dim3(D2 / 128, B * M / 128), blk, 0, stream>>>(src, qk_src_w, nullptr, qks, C, D2, 0);
    gemm_tn<false, true, false><<<dim3(Dd / 128, B * N / 128), blk, 0, stream>>>(x,   v_w,      nullptr, v,   C, Dd, 0);
    gemm_tn<false, true, false><<<dim3(Dd / 128, B * M / 128), blk, 0, stream>>>(src, v_src_w,  nullptr, vs,  C, Dd, 0);

    // Stage 2: bidirectional attention = two symmetric flash passes
    flash_bidir<<<dim3(N / 64, H, B), dim3(128), 0, stream>>>(qk,  qks, vs, y,  N, M, scale);
    flash_bidir<<<dim3(M / 64, H, B), dim3(128), 0, stream>>>(qks, qk,  v,  ys, M, N, scale);

    // Stage 3: output projections into concatenated d_out [B*N, 1536]
    gemm_tn<true, false, true><<<dim3(Dd / 128, B * N / 128), blk, 0, stream>>>(y,  proj_w,     proj_b,     out, Dd, 1536, 0);
    gemm_tn<true, false, true><<<dim3(Dd / 128, B * M / 128), blk, 0, stream>>>(ys, proj_src_w, proj_src_b, out, Dd, 1536, 768);
}
